// IdpGANBlock_30760555774597
// MI455X (gfx1250) — compile-verified, matrix operand order checked by reading, not run on hardware
//
#include <hip/hip_runtime.h>
#include <hip/hip_bf16.h>

// ---------------- problem constants ----------------
#define LSEQ 512
#define BBAT 8
#define EE   128
#define DD   384
#define HH   12
#define HDIM 32
#define E1D  64
#define FFD  512
#define CIN  (EE + E1D)        // 192
#define MROWS (LSEQ * BBAT)    // 4096
#define SCALE_QK 0.05103103630798287f  // 1/sqrt(384)

typedef __attribute__((ext_vector_type(16))) _Float16 v16h;
typedef __attribute__((ext_vector_type(8)))  float    v8f;

// ---------------- WMMA helpers ----------------
// D = A(16x32 f16) * B(32x16 f16) + C(16x16 f32)
__device__ __forceinline__ v8f wmma16(v16h a, v16h b, v8f c) {
  return __builtin_amdgcn_wmma_f32_16x16x32_f16(false, a, false, b, (short)0, c,
                                                false, false);
}

// Load one 16x32 f16 operand tile from row-major memory (row stride = ld
// elements). Works for the A operand (rows = M) and, because we always pass
// B pre-transposed (Bt is NxK row-major), for the B operand too (rows = N).
// CDNA5 16-bit A layout: lanes 0-15 hold K 0..7 & 16..23 of row m,
// lanes 16-31 hold K 8..15 & 24..31 of row m.
__device__ __forceinline__ v16h load_tile_f16(const _Float16* base, int ld) {
  const int lane = threadIdx.x & 31;
  const int m    = lane & 15;
  const int kb   = (lane >> 4) << 3;            // 0 or 8
  const _Float16* r0 = base + (size_t)m * ld + kb;
  v16h a;
  *((float4*)&a)       = *(const float4*)(r0);       // K kb..kb+7
  *(((float4*)&a) + 1) = *(const float4*)(r0 + 16);  // K kb+16..kb+23
  return a;
}

// Same tile load but source is fp32: convert to f16 on the fly.
__device__ __forceinline__ v16h load_tile_f32cvt(const float* base, int ld) {
  const int lane = threadIdx.x & 31;
  const int m    = lane & 15;
  const int kb   = (lane >> 4) << 3;
  const float* r0 = base + (size_t)m * ld + kb;
  float4 x0 = *(const float4*)(r0);
  float4 x1 = *(const float4*)(r0 + 4);
  float4 x2 = *(const float4*)(r0 + 16);
  float4 x3 = *(const float4*)(r0 + 20);
  v16h a;
  a[0]=(_Float16)x0.x; a[1]=(_Float16)x0.y; a[2]=(_Float16)x0.z; a[3]=(_Float16)x0.w;
  a[4]=(_Float16)x1.x; a[5]=(_Float16)x1.y; a[6]=(_Float16)x1.z; a[7]=(_Float16)x1.w;
  a[8]=(_Float16)x2.x; a[9]=(_Float16)x2.y; a[10]=(_Float16)x2.z; a[11]=(_Float16)x2.w;
  a[12]=(_Float16)x3.x; a[13]=(_Float16)x3.y; a[14]=(_Float16)x3.z; a[15]=(_Float16)x3.w;
  return a;
}

// reductions across the 16 lanes holding one C-tile row (xor masks < 16
// stay within each half-wave)
__device__ __forceinline__ float half16_max(float v) {
#pragma unroll
  for (int m = 1; m < 16; m <<= 1) v = fmaxf(v, __shfl_xor(v, m, 32));
  return v;
}
__device__ __forceinline__ float half16_sum(float v) {
#pragma unroll
  for (int m = 1; m < 16; m <<= 1) v += __shfl_xor(v, m, 32);
  return v;
}

// ---------------- small prep kernels ----------------
__global__ void pack_s_kernel(const float* __restrict__ s, _Float16* __restrict__ S16) {
  int i = blockIdx.x * blockDim.x + threadIdx.x;
  if (i < MROWS * EE) S16[i] = (_Float16)s[i];
}

// dst (C x R) f16 = transpose of src (R x C) fp32
__global__ void transpose_w_kernel(const float* __restrict__ src, _Float16* __restrict__ dst,
                                   int R, int C) {
  int i = blockIdx.x * blockDim.x + threadIdx.x;
  if (i < R * C) {
    int r = i / C, c = i % C;
    dst[(size_t)c * R + r] = (_Float16)src[i];
  }
}

// W2d (64x12) -> W2dT (16x64) f16, rows 12..15 zero padded
__global__ void w2dT_kernel(const float* __restrict__ W2d, _Float16* __restrict__ dst) {
  int i = blockIdx.x * blockDim.x + threadIdx.x;
  if (i < 16 * 64) {
    int h = i / 64, c = i % 64;
    dst[i] = (h < HH) ? (_Float16)W2d[c * HH + h] : (_Float16)0.f;
  }
}

// ---------------- QKV projection ----------------
// S16 (4096x128) @ W{q,k,v}^T -> q16/k16 (B,H,L,HD) f16 (q scaled), vT16 (B,H,HD,L)
__global__ void qkv_gemm_kernel(const _Float16* __restrict__ S16,
                                const _Float16* __restrict__ WqT,
                                const _Float16* __restrict__ WkT,
                                const _Float16* __restrict__ WvT,
                                _Float16* __restrict__ q16,
                                _Float16* __restrict__ k16,
                                _Float16* __restrict__ vT16) {
  const int mode = blockIdx.z;                // 0=q 1=k 2=v (uniform)
  const _Float16* WT = (mode == 0) ? WqT : (mode == 1) ? WkT : WvT;
  const int wave = threadIdx.x >> 5;
  const int tile = blockIdx.x * 4 + wave;     // 6144 tiles
  const int tn = tile % (DD / 16);
  const int tm = tile / (DD / 16);
  const _Float16* A  = S16 + (size_t)tm * 16 * EE;
  const _Float16* Bt = WT + (size_t)tn * 16 * EE;
  v8f c = {};
#pragma unroll
  for (int kk = 0; kk < EE; kk += 32)
    c = wmma16(load_tile_f16(A + kk, EE), load_tile_f16(Bt + kk, EE), c);
  const int lane = threadIdx.x & 31;
  const int n = lane & 15, moff = (lane >> 4) << 3;
  // branchless epilogue: uniform dst/scale, per-element index select
  const float scl = (mode == 0) ? SCALE_QK : 1.0f;
  _Float16* dst = (mode == 0) ? q16 : (mode == 1) ? k16 : vT16;
  const bool isv = (mode == 2);
#pragma unroll
  for (int r = 0; r < 8; ++r) {
    int row = tm * 16 + r + moff;             // l*8+b
    int col = tn * 16 + n;                    // h*32+hd
    int l = row >> 3, b = row & 7, h = col >> 5, hd = col & 31;
    size_t bh = (size_t)(b * HH + h);
    size_t idx = isv ? ((bh * HDIM + hd) * LSEQ + l)
                     : ((bh * LSEQ + l) * HDIM + hd);
    dst[idx] = (_Float16)(c[r] * scl);
  }
}

// ---------------- fused pair-bias + flash attention ----------------
// One workgroup (16 waves) owns (b, 16-row i-tile) for ALL 12 heads, so each
// p[b,i,j,:] row is read exactly once from HBM (no pbias materialization).
// Per 32-j step:
//   phase 1: wave w computes bias tile for i=i0+w, 32 j, 16 h via 4 WMMAs
//            (accumulator pre-loaded with b2d) into padded LDS
//   phase 2: waves 0..11 = head waves: score WMMA + bias add + online softmax,
//            LDS transpose of probs, 2 PV WMMAs
__global__ void attn_fused_kernel(const _Float16* __restrict__ q16,
                                  const _Float16* __restrict__ k16,
                                  const _Float16* __restrict__ vT16,
                                  const float* __restrict__ p,
                                  const _Float16* __restrict__ W2dT,
                                  const float* __restrict__ b2d,
                                  _Float16* __restrict__ O16) {
  __shared__ __align__(16) float bias_lds[16 * 32 * 17];   // [i][j(32)][h pad17]
  __shared__ __align__(16) _Float16 atile[16][16 * 32];    // per-head prob tiles
  const int wave = threadIdx.x >> 5;       // 0..15
  const int lane = threadIdx.x & 31;
  const int n = lane & 15, moff = (lane >> 4) << 3;
  const int b = blockIdx.x;                // 0..7
  const int i0 = blockIdx.y * 16;          // i-tile
  const int h = wave;                      // head id for wave < 12
  const size_t bh = (size_t)(b * HH + h);

  // per-head persistent state (only meaningful for wave < 12)
  v16h aq = {};
  if (wave < HH) aq = load_tile_f16(q16 + (bh * LSEQ + i0) * HDIM, HDIM);
  v8f oc0 = {}, oc1 = {};
  float rm[8], rl[8];
#pragma unroll
  for (int r = 0; r < 8; ++r) { rm[r] = -3.0e38f; rl[r] = 0.f; }

  const float bb = (n < HH) ? b2d[n] : 0.f;
  const float* Abase0 =
      p + (((size_t)b * LSEQ + (i0 + wave)) * LSEQ) * 64;   // row (b, i0+wave, j=0)
  _Float16* at = &atile[wave][0];

  for (int js = 0; js < LSEQ / 32; ++js) {
    const int j0 = js * 32;
    __syncthreads();   // previous iteration's bias_lds reads complete
    // ---- phase 1: pair bias for i = i0+wave, j0..j0+31, all heads ----
    {
      const float* Ab = Abase0 + (size_t)j0 * 64;
      __builtin_prefetch(Ab + 2048 + (lane << 4), 0, 1);  // stream next 32-j block
#pragma unroll
      for (int t = 0; t < 2; ++t) {
        v8f cb;
#pragma unroll
        for (int r = 0; r < 8; ++r) cb[r] = bb;
        cb = wmma16(load_tile_f32cvt(Ab + (size_t)t * 1024, 64),
                    load_tile_f16(W2dT, 64), cb);
        cb = wmma16(load_tile_f32cvt(Ab + (size_t)t * 1024 + 32, 64),
                    load_tile_f16(W2dT + 32, 64), cb);
#pragma unroll
        for (int r = 0; r < 8; ++r)
          bias_lds[(wave * 32 + t * 16 + r + moff) * 17 + n] = cb[r];
      }
    }
    __syncthreads();
    // ---- phase 2: attention step for head h = wave ----
    if (wave < HH) {
      v8f z0 = {}, z1 = {};
      v8f sc0 = wmma16(aq, load_tile_f16(k16 + (bh * LSEQ + j0) * HDIM, HDIM), z0);
      v8f sc1 = wmma16(aq, load_tile_f16(k16 + (bh * LSEQ + j0 + 16) * HDIM, HDIM), z1);
#pragma unroll
      for (int r = 0; r < 8; ++r) {
        int m = r + moff;
        float v0 = sc0[r] + bias_lds[(m * 32 + n) * 17 + h];
        float v1 = sc1[r] + bias_lds[(m * 32 + 16 + n) * 17 + h];
        float tmax = half16_max(fmaxf(v0, v1));
        float mnew = fmaxf(rm[r], tmax);
        float corr = __expf(rm[r] - mnew);
        float p0 = __expf(v0 - mnew);
        float p1 = __expf(v1 - mnew);
        rl[r] = rl[r] * corr + half16_sum(p0 + p1);
        rm[r] = mnew;
        oc0[r] *= corr;
        oc1[r] *= corr;
        at[m * 32 + n]      = (_Float16)p0;
        at[m * 32 + 16 + n] = (_Float16)p1;
      }
      __threadfence_block();   // order same-wave LDS transpose (DS in-order)
      v16h ap = load_tile_f16(at, 32);
      oc0 = wmma16(ap, load_tile_f16(vT16 + (bh * HDIM +  0) * LSEQ + j0, LSEQ), oc0);
      oc1 = wmma16(ap, load_tile_f16(vT16 + (bh * HDIM + 16) * LSEQ + j0, LSEQ), oc1);
    }
  }
  // ---- normalize + write O (row-major 4096 x 384), row = l*8+b ----
  if (wave < HH) {
#pragma unroll
    for (int r = 0; r < 8; ++r) {
      int l = i0 + r + moff;
      float iv = 1.0f / rl[r];
      size_t off = ((size_t)(l * BBAT + b)) * DD + h * HDIM;
      O16[off + n]      = (_Float16)(oc0[r] * iv);
      O16[off + 16 + n] = (_Float16)(oc1[r] * iv);
    }
  }
}

// ---------------- out projection + residual ----------------
__global__ void outproj_gemm_kernel(const _Float16* __restrict__ O16,
                                    const _Float16* __restrict__ WoT,
                                    const float* __restrict__ bo,
                                    const float* __restrict__ s,
                                    float* __restrict__ U) {
  const int wave = threadIdx.x >> 5;
  const int tile = blockIdx.x * 4 + wave;     // 2048 tiles
  const int tn = tile & 7, tm = tile >> 3;
  const _Float16* A  = O16 + (size_t)tm * 16 * DD;
  const _Float16* Bt = WoT + (size_t)tn * 16 * DD;
  v8f c = {};
#pragma unroll
  for (int kk = 0; kk < DD; kk += 32)
    c = wmma16(load_tile_f16(A + kk, DD), load_tile_f16(Bt + kk, DD), c);
  const int lane = threadIdx.x & 31;
  const int n = lane & 15, moff = (lane >> 4) << 3;
  int col = tn * 16 + n;
  float bias = bo[col];
#pragma unroll
  for (int r = 0; r < 8; ++r) {
    int row = tm * 16 + r + moff;
    U[(size_t)row * EE + col] = c[r] + bias + s[(size_t)row * EE + col];
  }
}

// ---------------- LayerNorm 1 (+ concat with x into Cin f16) ----------------
__global__ void ln1_kernel(const float* __restrict__ U, const float* __restrict__ g,
                           const float* __restrict__ be, const float* __restrict__ x,
                           float* __restrict__ sA, _Float16* __restrict__ Cin) {
  __shared__ float red[8];
  const int row = blockIdx.x, t = threadIdx.x;
  const int wave = t >> 5, lane = t & 31;
  float v = U[(size_t)row * EE + t];
  float s1 = v, s2 = v * v;
#pragma unroll
  for (int m = 1; m < 32; m <<= 1) { s1 += __shfl_xor(s1, m, 32); s2 += __shfl_xor(s2, m, 32); }
  if (lane == 0) { red[wave] = s1; red[4 + wave] = s2; }
  __syncthreads();
  float S1 = red[0] + red[1] + red[2] + red[3];
  float S2 = red[4] + red[5] + red[6] + red[7];
  float mean = S1 * (1.f / EE);
  float var = S2 * (1.f / EE) - mean * mean;
  float y = (v - mean) * rsqrtf(var + 1e-5f) * g[t] + be[t];
  sA[(size_t)row * EE + t] = y;
  Cin[(size_t)row * CIN + t] = (_Float16)y;
  if (t < E1D) Cin[(size_t)row * CIN + EE + t] = (_Float16)x[(size_t)row * E1D + t];
}

// ---------------- FFN1: relu(Cin @ W1 + b1) -> H16 ----------------
__global__ void ffn1_gemm_kernel(const _Float16* __restrict__ Cin,
                                 const _Float16* __restrict__ W1T,
                                 const float* __restrict__ b1,
                                 _Float16* __restrict__ H16) {
  const int wave = threadIdx.x >> 5;
  const int tile = blockIdx.x * 4 + wave;     // 8192 tiles
  const int tn = tile & 31, tm = tile >> 5;
  const _Float16* A  = Cin + (size_t)tm * 16 * CIN;
  const _Float16* Bt = W1T + (size_t)tn * 16 * CIN;
  v8f c = {};
#pragma unroll
  for (int kk = 0; kk < CIN; kk += 32)
    c = wmma16(load_tile_f16(A + kk, CIN), load_tile_f16(Bt + kk, CIN), c);
  const int lane = threadIdx.x & 31;
  const int n = lane & 15, moff = (lane >> 4) << 3;
  int col = tn * 16 + n;
  float bias = b1[col];
#pragma unroll
  for (int r = 0; r < 8; ++r) {
    int row = tm * 16 + r + moff;
    H16[(size_t)row * FFD + col] = (_Float16)fmaxf(c[r] + bias, 0.f);
  }
}

// ---------------- FFN2: H16 @ W2 + b2 + residual -> U2 ----------------
__global__ void ffn2_gemm_kernel(const _Float16* __restrict__ H16,
                                 const _Float16* __restrict__ W2T,
                                 const float* __restrict__ b2,
                                 const float* __restrict__ sA,
                                 float* __restrict__ U2) {
  const int wave = threadIdx.x >> 5;
  const int tile = blockIdx.x * 4 + wave;     // 2048 tiles
  const int tn = tile & 7, tm = tile >> 3;
  const _Float16* A  = H16 + (size_t)tm * 16 * FFD;
  const _Float16* Bt = W2T + (size_t)tn * 16 * FFD;
  v8f c = {};
#pragma unroll
  for (int kk = 0; kk < FFD; kk += 32)
    c = wmma16(load_tile_f16(A + kk, FFD), load_tile_f16(Bt + kk, FFD), c);
  const int lane = threadIdx.x & 31;
  const int n = lane & 15, moff = (lane >> 4) << 3;
  int col = tn * 16 + n;
  float bias = b2[col];
#pragma unroll
  for (int r = 0; r < 8; ++r) {
    int row = tm * 16 + r + moff;
    U2[(size_t)row * EE + col] = c[r] + bias + sA[(size_t)row * EE + col];
  }
}

// ---------------- LayerNorm 2 -> output ----------------
__global__ void ln2_kernel(const float* __restrict__ U2, const float* __restrict__ g,
                           const float* __restrict__ be, float* __restrict__ out) {
  __shared__ float red[8];
  const int row = blockIdx.x, t = threadIdx.x;
  const int wave = t >> 5, lane = t & 31;
  float v = U2[(size_t)row * EE + t];
  float s1 = v, s2 = v * v;
#pragma unroll
  for (int m = 1; m < 32; m <<= 1) { s1 += __shfl_xor(s1, m, 32); s2 += __shfl_xor(s2, m, 32); }
  if (lane == 0) { red[wave] = s1; red[4 + wave] = s2; }
  __syncthreads();
  float S1 = red[0] + red[1] + red[2] + red[3];
  float S2 = red[4] + red[5] + red[6] + red[7];
  float mean = S1 * (1.f / EE);
  float var = S2 * (1.f / EE) - mean * mean;
  out[(size_t)row * EE + t] = (v - mean) * rsqrtf(var + 1e-5f) * g[t] + be[t];
}

// ---------------- host launcher ----------------
extern "C" void kernel_launch(void* const* d_in, const int* in_sizes, int n_in,
                              void* d_out, int out_size, void* d_ws, size_t ws_size,
                              hipStream_t stream) {
  const float* s   = (const float*)d_in[0];
  const float* x   = (const float*)d_in[1];
  const float* p   = (const float*)d_in[2];
  const float* Wq  = (const float*)d_in[3];
  const float* Wk  = (const float*)d_in[4];
  const float* Wv  = (const float*)d_in[5];
  const float* Wo  = (const float*)d_in[6];
  const float* bo  = (const float*)d_in[7];
  const float* W2d = (const float*)d_in[8];
  const float* b2d = (const float*)d_in[9];
  const float* W1  = (const float*)d_in[10];
  const float* b1  = (const float*)d_in[11];
  const float* W2  = (const float*)d_in[12];
  const float* b2  = (const float*)d_in[13];
  const float* g1  = (const float*)d_in[14];
  const float* be1 = (const float*)d_in[15];
  const float* g2  = (const float*)d_in[16];
  const float* be2 = (const float*)d_in[17];

  char* w = (char*)d_ws;
  auto carve = [&](size_t bytes) {
    void* r = (void*)w;
    w += (bytes + 255) & ~(size_t)255;
    return r;
  };
  _Float16* S16  = (_Float16*)carve((size_t)MROWS * EE * 2);
  _Float16* WqT  = (_Float16*)carve((size_t)DD * EE * 2);
  _Float16* WkT  = (_Float16*)carve((size_t)DD * EE * 2);
  _Float16* WvT  = (_Float16*)carve((size_t)DD * EE * 2);
  _Float16* WoT  = (_Float16*)carve((size_t)EE * DD * 2);
  _Float16* W2dT = (_Float16*)carve((size_t)16 * 64 * 2);
  _Float16* W1T  = (_Float16*)carve((size_t)FFD * CIN * 2);
  _Float16* W2T  = (_Float16*)carve((size_t)EE * FFD * 2);
  _Float16* q16  = (_Float16*)carve((size_t)BBAT * HH * LSEQ * HDIM * 2);
  _Float16* k16  = (_Float16*)carve((size_t)BBAT * HH * LSEQ * HDIM * 2);
  _Float16* vT16 = (_Float16*)carve((size_t)BBAT * HH * HDIM * LSEQ * 2);
  _Float16* O16  = (_Float16*)carve((size_t)MROWS * DD * 2);
  float*    U    = (float*)carve((size_t)MROWS * EE * 4);
  float*    sA   = (float*)carve((size_t)MROWS * EE * 4);
  _Float16* Cin  = (_Float16*)carve((size_t)MROWS * CIN * 2);
  _Float16* H16  = (_Float16*)carve((size_t)MROWS * FFD * 2);
  float*    U2   = (float*)carve((size_t)MROWS * EE * 4);

  pack_s_kernel<<<(MROWS * EE + 255) / 256, 256, 0, stream>>>(s, S16);
  transpose_w_kernel<<<(EE * DD + 255) / 256, 256, 0, stream>>>(Wq, WqT, EE, DD);
  transpose_w_kernel<<<(EE * DD + 255) / 256, 256, 0, stream>>>(Wk, WkT, EE, DD);
  transpose_w_kernel<<<(EE * DD + 255) / 256, 256, 0, stream>>>(Wv, WvT, EE, DD);
  transpose_w_kernel<<<(DD * EE + 255) / 256, 256, 0, stream>>>(Wo, WoT, DD, EE);
  transpose_w_kernel<<<(CIN * FFD + 255) / 256, 256, 0, stream>>>(W1, W1T, CIN, FFD);
  transpose_w_kernel<<<(FFD * EE + 255) / 256, 256, 0, stream>>>(W2, W2T, FFD, EE);
  w2dT_kernel<<<(16 * 64 + 255) / 256, 256, 0, stream>>>(W2d, W2dT);

  // QKV: 6144 tiles per matrix / 4 waves
  qkv_gemm_kernel<<<dim3(1536, 1, 3), 128, 0, stream>>>(S16, WqT, WkT, WvT, q16, k16, vT16);
  // fused pair-bias + attention: (b, i-tile) grid, 16 waves each
  attn_fused_kernel<<<dim3(BBAT, LSEQ / 16), 512, 0, stream>>>(q16, k16, vT16, p, W2dT,
                                                               b2d, O16);
  // out projection + residual: 2048 tiles / 4
  outproj_gemm_kernel<<<512, 128, 0, stream>>>(O16, WoT, bo, s, U);
  ln1_kernel<<<MROWS, 128, 0, stream>>>(U, g1, be1, x, sA, Cin);
  // FFN1: 8192 tiles / 4
  ffn1_gemm_kernel<<<2048, 128, 0, stream>>>(Cin, W1T, b1, H16);
  // FFN2: 2048 tiles / 4
  ffn2_gemm_kernel<<<512, 128, 0, stream>>>(H16, W2T, b2, sA, U2);
  ln2_kernel<<<MROWS, 128, 0, stream>>>(U2, g2, be2, (float*)d_out);
}
